// Model_39676907884408
// MI455X (gfx1250) — matrix operand order repaired and checked by reading, not compiled, not run on hardware
//
#include <hip/hip_runtime.h>
#include <hip/hip_bf16.h>

#define B_  4
#define H_  16
#define SQ_ 2048
#define SKV_ 2048
#define D_  128
#define KV_TILE 64
#define KSTRIDE 136   // 128 + 8 pad (bf16 elems) -> 272B rows, 16B aligned
#define VSTRIDE 72    // 64 + 8 pad -> 144B rows, 16B aligned
#define PSTRIDE 72

typedef __attribute__((ext_vector_type(16))) __bf16 v16bf;
typedef __attribute__((ext_vector_type(8)))  float  v8f;

// 16 contiguous bf16 (B-fragment): elements 0..15 at p..p+15
__device__ __forceinline__ v16bf lds_frag_contig(const __bf16* p) {
    union { uint4 u[2]; v16bf v; } t;
    t.u[0] = *(const uint4*)(p);
    t.u[1] = *(const uint4*)(p + 8);
    return t.v;
}
// A-fragment: elements 0..7 at p, elements 8..15 at p+16
__device__ __forceinline__ v16bf lds_frag_split(const __bf16* p) {
    union { uint4 u[2]; v16bf v; } t;
    t.u[0] = *(const uint4*)(p);
    t.u[1] = *(const uint4*)(p + 16);
    return t.v;
}

__global__ __launch_bounds__(256) void fa_fwd_bf16wmma(
        const float* __restrict__ Q, const float* __restrict__ K,
        const float* __restrict__ V, float* __restrict__ O) {
    __shared__ __align__(16) __bf16 Klds[KV_TILE][KSTRIDE];
    __shared__ __align__(16) __bf16 Vt[D_][VSTRIDE];          // transposed V tile
    __shared__ __align__(16) __bf16 Plds[8][16][PSTRIDE];     // per-wave P scratch

    const int tid  = threadIdx.x;
    const int wave = tid >> 5;
    const int lane = tid & 31;
    const int half = lane >> 4;   // 0: rows/K-lo half, 1: rows/K-hi half
    const int l16  = lane & 15;

    const int bh = blockIdx.y;                      // 0..63
    const int q0 = blockIdx.x * 128 + wave * 16;    // 16 q rows per wave

    const float* Qb = Q + (size_t)bh * SQ_ * D_;
    const float* Kb = K + (size_t)bh * SKV_ * D_;
    const float* Vb = V + (size_t)bh * SKV_ * D_;
    float*       Ob = O + (size_t)bh * SQ_ * D_;

    // softmax in base 2: fold log2(e)/sqrt(D) into Q before bf16 conversion
    const float QSCALE = 1.4426950408889634f * 0.08838834764831845f;

    // ---- load Q tile once as 4 bf16 A-fragments (K-dim = d, 32 each) ----
    v16bf qfrag[4];
    {
        const float* qrowp = Qb + (size_t)(q0 + l16) * D_;
#pragma unroll
        for (int f = 0; f < 4; ++f) {
            const float* p0 = qrowp + f * 32 + half * 8;  // K = f*32+half*8 .. +7
            const float* p1 = p0 + 16;                    // K = f*32+16+half*8 ..
            v16bf t;
#pragma unroll
            for (int i = 0; i < 8; ++i) t[i]     = (__bf16)(p0[i] * QSCALE);
#pragma unroll
            for (int i = 0; i < 8; ++i) t[8 + i] = (__bf16)(p1[i] * QSCALE);
            qfrag[f] = t;
        }
    }

    // ---- accumulators ----
    const v8f zero8 = {0.f,0.f,0.f,0.f,0.f,0.f,0.f,0.f};
    v8f  o[8];
#pragma unroll
    for (int dt = 0; dt < 8; ++dt) o[dt] = zero8;
    float m[8], l[8];
#pragma unroll
    for (int r = 0; r < 8; ++r) { m[r] = -1e30f; l[r] = 0.f; }

    for (int kv0 = 0; kv0 < SKV_; kv0 += KV_TILE) {
        // ---- cooperative stage: K (row-major bf16) and V (transposed bf16) ----
        {
            const float4* Ksrc = (const float4*)(Kb + (size_t)kv0 * D_);
            const float4* Vsrc = (const float4*)(Vb + (size_t)kv0 * D_);
#pragma unroll
            for (int j = 0; j < 8; ++j) {
                int idx = j * 256 + tid;        // 0..2047 float4s of the 64x128 tile
                int row = idx >> 5;             // kv row in tile
                int c   = (idx & 31) * 4;       // d column
                float4 kk = Ksrc[idx];
                float4 vv = Vsrc[idx];
                Klds[row][c + 0] = (__bf16)kk.x;
                Klds[row][c + 1] = (__bf16)kk.y;
                Klds[row][c + 2] = (__bf16)kk.z;
                Klds[row][c + 3] = (__bf16)kk.w;
                Vt[c + 0][row] = (__bf16)vv.x;
                Vt[c + 1][row] = (__bf16)vv.y;
                Vt[c + 2][row] = (__bf16)vv.z;
                Vt[c + 3][row] = (__bf16)vv.w;
            }
        }
        __syncthreads();

        // ---- S = Q * K^T  (4 n-tiles of 16 kv columns, K-dim = d = 128) ----
        v8f s[4];
#pragma unroll
        for (int nt = 0; nt < 4; ++nt) {
            v8f acc = zero8;
#pragma unroll
            for (int f = 0; f < 4; ++f) {
                v16bf bk = lds_frag_split(&Klds[nt * 16 + l16][f * 32 + half * 8]);
                acc = __builtin_amdgcn_wmma_f32_16x16x32_bf16(
                        false, qfrag[f], false, bk, (short)0, acc, false, false);
            }
            s[nt] = acc;
        }

        // ---- online softmax (rows striped: lane-half h holds rows h*8+r) ----
#pragma unroll
        for (int r = 0; r < 8; ++r) {
            float mloc = fmaxf(fmaxf(s[0][r], s[1][r]), fmaxf(s[2][r], s[3][r]));
#pragma unroll
            for (int off = 1; off < 16; off <<= 1)
                mloc = fmaxf(mloc, __shfl_xor(mloc, off, 32));
            float mnew = fmaxf(m[r], mloc);
            float corr = exp2f(m[r] - mnew);
            m[r] = mnew;
            float lloc = 0.f;
#pragma unroll
            for (int nt = 0; nt < 4; ++nt) {
                float p = exp2f(s[nt][r] - mnew);
                s[nt][r] = p;
                lloc += p;
            }
#pragma unroll
            for (int off = 1; off < 16; off <<= 1)
                lloc += __shfl_xor(lloc, off, 32);
            l[r] = l[r] * corr + lloc;
#pragma unroll
            for (int dt = 0; dt < 8; ++dt) o[dt][r] *= corr;
        }

        // ---- transpose P (C-layout -> A-layout) through per-wave LDS ----
#pragma unroll
        for (int nt = 0; nt < 4; ++nt)
#pragma unroll
            for (int r = 0; r < 8; ++r)
                Plds[wave][half * 8 + r][nt * 16 + l16] = (__bf16)s[nt][r];
        asm volatile("s_wait_dscnt 0" ::: "memory");

        v16bf pfrag[2];
#pragma unroll
        for (int kf = 0; kf < 2; ++kf)
            pfrag[kf] = lds_frag_split(&Plds[wave][l16][kf * 32 + half * 8]);

        // ---- O += P * V  (8 d-tiles, K-dim = kv = 64) ----
#pragma unroll
        for (int dt = 0; dt < 8; ++dt) {
            v8f acc = o[dt];
#pragma unroll
            for (int kf = 0; kf < 2; ++kf) {
                v16bf bv = lds_frag_split(&Vt[dt * 16 + l16][kf * 32 + half * 8]);
                acc = __builtin_amdgcn_wmma_f32_16x16x32_bf16(
                        false, pfrag[kf], false, bv, (short)0, acc, false, false);
            }
            o[dt] = acc;
        }
        __syncthreads();
    }

    // ---- epilogue: O / l, fp32 row-major store ----
#pragma unroll
    for (int r = 0; r < 8; ++r) {
        float inv = 1.0f / l[r];
        int row = q0 + half * 8 + r;
        float* orow = Ob + (size_t)row * D_;
#pragma unroll
        for (int dt = 0; dt < 8; ++dt)
            orow[dt * 16 + l16] = o[dt][r] * inv;
    }
}

extern "C" void kernel_launch(void* const* d_in, const int* in_sizes, int n_in,
                              void* d_out, int out_size, void* d_ws, size_t ws_size,
                              hipStream_t stream) {
    const float* q = (const float*)d_in[0];
    const float* k = (const float*)d_in[1];
    const float* v = (const float*)d_in[2];
    float* out = (float*)d_out;
    dim3 grid(SQ_ / 128, B_ * H_);
    fa_fwd_bf16wmma<<<grid, dim3(256), 0, stream>>>(q, k, v, out);
}
